// WindowProcessor_13941463843193
// MI455X (gfx1250) — matrix operand order repaired and checked by reading, not compiled, not run on hardware
//
#include <hip/hip_runtime.h>
#include <hip/hip_bf16.h>

// ---------------------------------------------------------------------------
// Types for CDNA5 WMMA (wave32): v_wmma_f32_16x16x32_bf16
// ---------------------------------------------------------------------------
typedef __bf16 bf16;
typedef __attribute__((ext_vector_type(8)))  bf16  bf16x8;
typedef __attribute__((ext_vector_type(16))) bf16  bf16x16;
typedef __attribute__((ext_vector_type(8)))  float f32x8;
typedef __attribute__((ext_vector_type(4)))  int   v4i;

union Frag { bf16x16 v; bf16x8 h[2]; };

// Optional CDNA5 async global->LDS path (guarded so a missing builtin cannot
// break the build; fallback is a synchronous VGPR bounce).
#if defined(__has_builtin)
#if __has_builtin(__builtin_amdgcn_global_load_async_to_lds_b128) && \
    __has_builtin(__builtin_amdgcn_s_wait_asynccnt)
#define USE_ASYNC_LDS 1
#endif
#endif
#ifndef USE_ASYNC_LDS
#define USE_ASYNC_LDS 0
#endif

__device__ __forceinline__ void async_wait_all() {
#if USE_ASYNC_LDS
    __builtin_amdgcn_s_wait_asynccnt(0);
#endif
}

// f32 -> bf16 round-to-nearest-even (storage type; avoids scalar cast codegen risk)
__device__ __forceinline__ bf16 f2bf(float f) {
    union { float f; unsigned u; } a; a.f = f;
    unsigned r = (a.u + 0x7FFFu + ((a.u >> 16) & 1u)) >> 16;
    union { unsigned short s; bf16 b; } o; o.s = (unsigned short)r;
    return o.b;
}

// A-matrix fragment (16x32 bf16, row-major source [rows, ld]):
// lane L holds row (lane&15); K elems {ko..ko+7, ko+16..ko+23}, ko = (lane>>4)*8
__device__ __forceinline__ bf16x16 fragA(const bf16* base, int ld, int row0, int k0) {
    int lane = threadIdx.x & 31;
    const bf16* p = base + (size_t)(row0 + (lane & 15)) * ld + k0 + ((lane >> 4) << 3);
    Frag f;
    f.h[0] = *(const bf16x8*)(p);
    f.h[1] = *(const bf16x8*)(p + 16);
    return f.v;
}

// B-matrix fragment (32x16 bf16) built from row-major W[N,K]:
// lane L holds col n=(lane&15); K elems {ko..ko+7, ko+16..ko+23}, ko = (lane>>4)*8
__device__ __forceinline__ bf16x16 fragB(const bf16* base, int ld, int n0, int k0) {
    int lane = threadIdx.x & 31;
    const bf16* p = base + (size_t)(n0 + (lane & 15)) * ld + k0 + ((lane >> 4) << 3);
    Frag f;
    f.h[0] = *(const bf16x8*)(p);
    f.h[1] = *(const bf16x8*)(p + 16);
    return f.v;
}

#define WMMA_BF16(A, B, Cacc) \
    __builtin_amdgcn_wmma_f32_16x16x32_bf16(false, (A), false, (B), (short)0, (Cacc), false, false)

// ---------------------------------------------------------------------------
// Generic GEMM: out[M,N] = act(A[M,K](bf16) @ W[N,K]^T(bf16) + bias [+ R]),
// f32 and/or bf16 outputs. 256 thr = 8 waves; wave tile 32x64; block 128x128.
// W tile (128x32/k-step) staged in LDS, double buffered (async if available).
// Requires M%128==0, N%128==0, K%32==0.
// ---------------------------------------------------------------------------
__device__ __forceinline__ void stage_w(const bf16* __restrict__ W, int K,
                                        int colBase, int k0, bf16* dstBuf) {
    int tid = threadIdx.x;
    #pragma unroll
    for (int rep = 0; rep < 2; ++rep) {
        int idx = tid + rep * 256;                 // 0..511 -> 128 rows x 4 chunks
        int r  = idx >> 2;
        int ch = (idx & 3) << 3;
        const bf16* src = W + (size_t)(colBase + r) * K + k0 + ch;
        bf16* dst = dstBuf + r * 32 + ch;
#if USE_ASYNC_LDS
        __builtin_amdgcn_global_load_async_to_lds_b128(
            (__attribute__((address_space(1))) v4i*)(void*)src,
            (__attribute__((address_space(3))) v4i*)(void*)dst, 0, 0);
#else
        *(bf16x8*)dst = *(const bf16x8*)src;
#endif
    }
}

__global__ __launch_bounds__(256)
void gemm_bf16_kernel(const bf16* __restrict__ A, const bf16* __restrict__ W,
                      const float* __restrict__ bias, const float* __restrict__ R,
                      float* __restrict__ outF, bf16* __restrict__ outB,
                      int M, int N, int K, int relu)
{
    __shared__ bf16 sW[2][128 * 32];               // 16 KB double-buffered W tile

    int wave = threadIdx.x >> 5;
    int lane = threadIdx.x & 31;
    int row0 = blockIdx.y * 128 + (wave & 3) * 32;
    int colBase = blockIdx.x * 128;
    int colLoc  = (wave >> 2) * 64;                // 0 or 64 within block tile

    f32x8 acc[2][4] = {};

    stage_w(W, K, colBase, 0, sW[0]);
    async_wait_all();
    __syncthreads();

    int buf = 0;
    for (int k0 = 0; k0 < K; k0 += 32) {
        if (k0 + 32 < K) stage_w(W, K, colBase, k0 + 32, sW[buf ^ 1]);
        if (k0 + 64 < K) {
            // far prefetch of the A tile two k-steps ahead (global_prefetch_b8)
            __builtin_prefetch(A + (size_t)(row0 + lane) * K + k0 + 64, 0, 1);
        }
        bf16x16 a0 = fragA(A, K, row0,      k0);
        bf16x16 a1 = fragA(A, K, row0 + 16, k0);
        #pragma unroll
        for (int j = 0; j < 4; ++j) {
            bf16x16 b = fragB(sW[buf], 32, colLoc + j * 16, 0);
            acc[0][j] = WMMA_BF16(a0, b, acc[0][j]);
            acc[1][j] = WMMA_BF16(a1, b, acc[1][j]);
        }
        async_wait_all();
        __syncthreads();
        buf ^= 1;
    }

    int nbase = lane & 15;
    int mhalf = (lane >> 4) << 3;
    #pragma unroll
    for (int i = 0; i < 2; ++i) {
        #pragma unroll
        for (int j = 0; j < 4; ++j) {
            int tr = row0 + i * 16 + mhalf;
            int tc = colBase + colLoc + j * 16 + nbase;
            float bv = bias ? bias[tc] : 0.0f;
            #pragma unroll
            for (int g = 0; g < 8; ++g) {
                size_t idx = (size_t)(tr + g) * N + tc;
                float v = acc[i][j][g] + bv;
                if (R)    v += R[idx];
                if (relu) v = fmaxf(v, 0.0f);
                if (outF) outF[idx] = v;
                if (outB) outB[idx] = f2bf(v);
            }
        }
    }
}

// ---------------------------------------------------------------------------
// Fused attention: one block per (group, head, 32-query-block). dh = 32.
// mode 0: group=(b,win), S=hw=256, stride=1.  mode 1: group=(b,pos), S=Nv=64, stride=hw.
// LDS: 32KB f32 scores (aliased by bf16 probs) + 16.9KB V^T + 1KB reductions.
// ---------------------------------------------------------------------------
#define QB 32
__global__ __launch_bounds__(256)
void attn_kernel(const bf16* __restrict__ Q, const bf16* __restrict__ K,
                 const bf16* __restrict__ V, bf16* __restrict__ O,
                 int S, int mode, int hw, int Nvhw)
{
    __shared__ float sS[QB][256];       // scores; later aliased as bf16 probs
    __shared__ bf16  sVt[32][264];      // V^T (dh x S), row stride 528B (16B-aligned)
    __shared__ float sRed[QB][8];
    bf16* sPa = (bf16*)&sS[0][0];       // probs [QB][256] bf16

    const int Cd = 256;
    int nQB  = S / QB;
    int bx   = blockIdx.x;
    int qb   = bx % nQB;
    int head = (bx / nQB) & 7;
    int gid  = bx / (nQB * 8);

    int base, stride;
    if (mode == 0) { base = gid * S; stride = 1; }
    else           { base = (gid / hw) * Nvhw + (gid % hw); stride = hw; }

    int tid = threadIdx.x, lane = tid & 31, wave = tid >> 5;
    int hoff  = head * 32;
    int lr    = lane & 15;
    int koA   = (lane >> 4) << 3;
    int koB   = (lane >> 4) << 4;
    int mhalf = (lane >> 4) << 3;

    // Stage V^T (dh x S) into LDS (consumed after the pre-P@V barrier)
    for (int idx = tid; idx < S * 32; idx += 256) {
        int s = idx >> 5, d = idx & 31;
        sVt[d][s] = V[(size_t)(base + s * stride) * Cd + hoff + d];
    }

    // Scores = Q K^T / sqrt(dh): 2 m-tiles x (S/16) n-tiles over 8 waves
    int q0 = qb * QB;
    {
        Frag fa[2];
        #pragma unroll
        for (int mt = 0; mt < 2; ++mt) {
            const bf16* pq = Q + (size_t)(base + (q0 + mt * 16 + lr) * stride) * Cd + hoff + koA;
            fa[mt].h[0] = *(const bf16x8*)pq;
            fa[mt].h[1] = *(const bf16x8*)(pq + 16);
        }
        const float scale = 0.17677669529663687f; // 1/sqrt(32)
        int nTiles = (S / 16) * 2;
        for (int t = wave; t < nTiles; t += 8) {
            int mt = t & 1, nt = t >> 1;
            const bf16* pk = K + (size_t)(base + (nt * 16 + lr) * stride) * Cd + hoff + koA;
            Frag fb; fb.h[0] = *(const bf16x8*)pk; fb.h[1] = *(const bf16x8*)(pk + 16);
            f32x8 c = {};
            c = WMMA_BF16(fa[mt].v, fb.v, c);
            int n  = nt * 16 + lr;
            int m0 = mt * 16 + mhalf;
            #pragma unroll
            for (int g = 0; g < 8; ++g) sS[m0 + g][n] = c[g] * scale;
        }
    }
    __syncthreads();

    // Row softmax over S (8 threads per row); probs -> bf16 aliased over sS.
    {
        int r = tid >> 3, p = tid & 7;
        int cpt  = S >> 3;              // 32 (S=256) or 8 (S=64)
        int cbeg = p * cpt;
        float mx = -3.4e38f;
        for (int c = 0; c < cpt; ++c) mx = fmaxf(mx, sS[r][cbeg + c]);
        sRed[r][p] = mx;
        __syncthreads();
        if (p == 0) { float m2 = sRed[r][0]; for (int i = 1; i < 8; ++i) m2 = fmaxf(m2, sRed[r][i]); sRed[r][0] = m2; }
        __syncthreads();
        float m = sRed[r][0];
        float ev[32];
        float sum = 0.0f;
        #pragma unroll
        for (int c = 0; c < 32; ++c) {
            if (c < cpt) { float e = __expf(sS[r][cbeg + c] - m); ev[c] = e; sum += e; }
        }
        __syncthreads();
        sRed[r][p] = sum;
        __syncthreads();
        if (p == 0) { float s2 = 0.0f; for (int i = 0; i < 8; ++i) s2 += sRed[r][i]; sRed[r][0] = 1.0f / s2; }
        __syncthreads();
        float inv = sRed[r][0];
        __syncthreads();   // every thread's f32 reads are in ev[]; aliased writes now safe
        #pragma unroll
        for (int c = 0; c < 32; ++c) {
            if (c < cpt) sPa[r * 256 + cbeg + c] = f2bf(ev[c] * inv);
        }
    }
    __syncthreads();

    // Out = P @ V : 2 m-tiles x 2 dh-tiles on waves 0..3
    if (wave < 4) {
        int mt = wave & 1, nt = wave >> 1;
        f32x8 c = {};
        for (int ks = 0; ks < S; ks += 32) {
            const bf16* pp = sPa + (mt * 16 + lr) * 256 + ks + koA;
            Frag fa; fa.h[0] = *(const bf16x8*)pp; fa.h[1] = *(const bf16x8*)(pp + 16);
            const bf16* pv = &sVt[nt * 16 + lr][ks + koA];
            Frag fb; fb.h[0] = *(const bf16x8*)pv; fb.h[1] = *(const bf16x8*)(pv + 16);
            c = WMMA_BF16(fa.v, fb.v, c);
        }
        int n  = hoff + nt * 16 + lr;
        int m0 = q0 + mt * 16 + mhalf;
        #pragma unroll
        for (int g = 0; g < 8; ++g)
            O[(size_t)(base + (m0 + g) * stride) * Cd + n] = f2bf(c[g]);
    }
}

// ---------------------------------------------------------------------------
// Per-token LayerNorm over C=256; optional f32/bf16/(+pe bf16) outputs.
// ---------------------------------------------------------------------------
__global__ __launch_bounds__(256)
void ln_kernel(const float* __restrict__ X, const float* __restrict__ g,
               const float* __restrict__ be, float* __restrict__ outF,
               bf16* __restrict__ outB, bf16* __restrict__ outPE,
               const float* __restrict__ pe, int Nvhw)
{
    __shared__ float red[256];
    int t = blockIdx.x;
    int c = threadIdx.x;
    size_t idx = (size_t)t * 256 + c;
    float x = X[idx];
    red[c] = x; __syncthreads();
    for (int s = 128; s > 0; s >>= 1) { if (c < s) red[c] += red[c + s]; __syncthreads(); }
    float mean = red[0] * (1.0f / 256.0f);
    __syncthreads();
    float d = x - mean;
    red[c] = d * d; __syncthreads();
    for (int s = 128; s > 0; s >>= 1) { if (c < s) red[c] += red[c + s]; __syncthreads(); }
    float var = red[0] * (1.0f / 256.0f);
    float y = d * rsqrtf(var + 1e-5f) * g[c] + be[c];
    if (outF) outF[idx] = y;
    if (outB) outB[idx] = f2bf(y);
    if (outPE) outPE[idx] = f2bf(y + pe[(size_t)(t % Nvhw) * 256 + c]);
}

// ---------------------------------------------------------------------------
// pe[v*hw+s, c] = glob[c, hf, wf] + relMLP(coords(s))[c]   (grid: hw blocks)
// ---------------------------------------------------------------------------
__global__ __launch_bounds__(256)
void pe_kernel(const float* __restrict__ glob, const float* __restrict__ relW1,
               const float* __restrict__ relb1, const float* __restrict__ relW2,
               const float* __restrict__ relb2, float* __restrict__ pe)
{
    __shared__ float hid[64];
    int s  = blockIdx.x;            // 0..255
    int iy = s >> 4, ix = s & 15;
    float cx = (float)ix * (1.0f / 15.0f);
    float cy = (float)iy * (1.0f / 15.0f);
    int c = threadIdx.x;
    if (c < 64) hid[c] = fmaxf(cx * relW1[c * 2 + 0] + cy * relW1[c * 2 + 1] + relb1[c], 0.0f);
    __syncthreads();
    float rv = relb2[c];
    for (int j = 0; j < 64; ++j) rv += relW2[c * 64 + j] * hid[j];
    for (int v = 0; v < 64; ++v) {
        int hf = (v >> 3) * 16 + iy;
        int wf = (v & 7) * 16 + ix;
        pe[(size_t)(v * 256 + s) * 256 + c] = glob[((size_t)c * 128 + hf) * 128 + wf] + rv;
    }
}

// Window partition: X32/Xbf = windows of backbone; XQK = bf16(X + pe)
__global__ __launch_bounds__(256)
void winpart_kernel(const float* __restrict__ bb, const float* __restrict__ pe,
                    float* __restrict__ X32, bf16* __restrict__ Xbf, bf16* __restrict__ XQK)
{
    int t = blockIdx.x;            // 0..B*Nv*hw-1
    int c = threadIdx.x;
    int b = t >> 14;
    int r = t & 16383;
    int v = r >> 8, s = r & 255;
    int hf = (v >> 3) * 16 + (s >> 4);
    int wf = (v & 7) * 16 + (s & 15);
    float x = bb[(((size_t)b * 256 + c) * 128 + hf) * 128 + wf];
    size_t idx = (size_t)t * 256 + c;
    X32[idx] = x;
    Xbf[idx] = f2bf(x);
    XQK[idx] = f2bf(x + pe[(size_t)r * 256 + c]);
}

// Scatter windows back + residual add into d_out
__global__ __launch_bounds__(256)
void final_kernel(const float* __restrict__ bb, const float* __restrict__ X32,
                  float* __restrict__ out)
{
    int t = blockIdx.x;
    int c = threadIdx.x;
    int b = t >> 14;
    int r = t & 16383;
    int v = r >> 8, s = r & 255;
    int hf = (v >> 3) * 16 + (s >> 4);
    int wf = (v & 7) * 16 + (s & 15);
    size_t o = (((size_t)b * 256 + c) * 128 + hf) * 128 + wf;
    out[o] = bb[o] + X32[(size_t)t * 256 + c];
}

// defe_mask: 16x16 block max-pool > 0 (written as 1.0/0.0 floats)
__global__ void mask_kernel(const float* __restrict__ defe, float* __restrict__ out)
{
    int t = threadIdx.x;
    if (t >= 128) return;
    int b = t >> 6, v = t & 63;
    int ny = v >> 3, nx = v & 7;
    float mx = -3.4e38f;
    for (int iy = 0; iy < 16; ++iy)
        for (int ix = 0; ix < 16; ++ix)
            mx = fmaxf(mx, defe[((size_t)b * 128 + (ny * 16 + iy)) * 128 + (nx * 16 + ix)]);
    out[t] = (mx > 0.0f) ? 1.0f : 0.0f;
}

__global__ void cvt_kernel(const float* __restrict__ in, bf16* __restrict__ out, int n)
{
    int i = blockIdx.x * 256 + threadIdx.x;
    if (i < n) out[i] = f2bf(in[i]);
}

// ---------------------------------------------------------------------------
// Host orchestration
// ---------------------------------------------------------------------------
extern "C" void kernel_launch(void* const* d_in, const int* in_sizes, int n_in,
                              void* d_out, int out_size, void* d_ws, size_t ws_size,
                              hipStream_t stream)
{
    (void)in_sizes; (void)n_in; (void)out_size; (void)ws_size;

    const float* backbone = (const float*)d_in[0];
    const float* defe     = (const float*)d_in[1];
    const float* glob     = (const float*)d_in[2];
    const float* Wqkv     = (const float*)d_in[3];
    const float* bqkv     = (const float*)d_in[4];
    const float* Wo       = (const float*)d_in[5];
    const float* bo       = (const float*)d_in[6];
    const float* W1       = (const float*)d_in[7];
    const float* b1       = (const float*)d_in[8];
    const float* W2       = (const float*)d_in[9];
    const float* b2       = (const float*)d_in[10];
    const float* g1       = (const float*)d_in[11];
    const float* be1      = (const float*)d_in[12];
    const float* g2       = (const float*)d_in[13];
    const float* be2      = (const float*)d_in[14];
    const float* relW1    = (const float*)d_in[15];
    const float* relb1    = (const float*)d_in[16];
    const float* relW2    = (const float*)d_in[17];
    const float* relb2    = (const float*)d_in[18];

    const int Cd = 256, hw = 256, Nvhw = 16384;      // Nv*hw
    const int M = 32768;                              // B*Nv*hw tokens
    const int DFF = 1024;

    // workspace carve-out
    char* ws = (char*)d_ws;
    size_t off = 0;
    auto carve = [&](size_t bytes) -> void* {
        void* p = ws + off;
        off = (off + bytes + 255) & ~(size_t)255;
        return p;
    };
    float* pe   = (float*)carve((size_t)Nvhw * Cd * 4);
    float* X32  = (float*)carve((size_t)M * Cd * 4);
    bf16*  Xbf  = (bf16*) carve((size_t)M * Cd * 2);
    bf16*  XQK  = (bf16*) carve((size_t)M * Cd * 2);
    bf16*  Qb   = (bf16*) carve((size_t)M * Cd * 2);
    bf16*  Kb   = (bf16*) carve((size_t)M * Cd * 2);
    bf16*  Vb   = (bf16*) carve((size_t)M * Cd * 2);
    bf16*  Ab   = (bf16*) carve((size_t)M * Cd * 2);
    float* T32  = (float*)carve((size_t)M * Cd * 4);
    float* S32  = (float*)carve((size_t)M * Cd * 4);
    bf16*  Sbf  = (bf16*) carve((size_t)M * Cd * 2);
    bf16*  Hbf  = (bf16*) carve((size_t)M * DFF * 2);
    bf16*  Wqkv_b = (bf16*)carve((size_t)768 * Cd * 2);
    bf16*  Wo_b   = (bf16*)carve((size_t)Cd * Cd * 2);
    bf16*  W1_b   = (bf16*)carve((size_t)DFF * Cd * 2);
    bf16*  W2_b   = (bf16*)carve((size_t)Cd * DFF * 2);

    // weight conversions
    cvt_kernel<<<(768 * Cd) / 256, 256, 0, stream>>>(Wqkv, Wqkv_b, 768 * Cd);
    cvt_kernel<<<(Cd * Cd) / 256, 256, 0, stream>>>(Wo, Wo_b, Cd * Cd);
    cvt_kernel<<<(DFF * Cd) / 256, 256, 0, stream>>>(W1, W1_b, DFF * Cd);
    cvt_kernel<<<(Cd * DFF) / 256, 256, 0, stream>>>(W2, W2_b, Cd * DFF);

    // positional embedding + window partition
    pe_kernel<<<hw, 256, 0, stream>>>(glob, relW1, relb1, relW2, relb2, pe);
    winpart_kernel<<<M, 256, 0, stream>>>(backbone, pe, X32, Xbf, XQK);

    for (int pass = 0; pass < 2; ++pass) {
        // QKV projections (bf16 GEMMs, wmma)
        dim3 g256(Cd / 128, M / 128);
        gemm_bf16_kernel<<<g256, 256, 0, stream>>>(XQK, Wqkv_b,            bqkv,       nullptr, nullptr, Qb, M, Cd, Cd, 0);
        gemm_bf16_kernel<<<g256, 256, 0, stream>>>(XQK, Wqkv_b + 256 * Cd, bqkv + 256, nullptr, nullptr, Kb, M, Cd, Cd, 0);
        gemm_bf16_kernel<<<g256, 256, 0, stream>>>(Xbf, Wqkv_b + 512 * Cd, bqkv + 512, nullptr, nullptr, Vb, M, Cd, Cd, 0);

        // attention: pass0 128 groups x S=256, pass1 512 groups x S=64
        int S = (pass == 0) ? 256 : 64;
        int nblk = (pass == 0) ? (2 * 64) * 8 * (256 / QB) : (2 * 256) * 8 * (64 / QB);
        attn_kernel<<<nblk, 256, 0, stream>>>(Qb, Kb, Vb, Ab, S, pass, hw, Nvhw);

        // out-proj + residual(X) -> T32 ; LN1 -> src
        gemm_bf16_kernel<<<g256, 256, 0, stream>>>(Ab, Wo_b, bo, X32, T32, nullptr, M, Cd, Cd, 0);
        ln_kernel<<<M, 256, 0, stream>>>(T32, g1, be1, S32, Sbf, nullptr, pe, Nvhw);

        // FFN
        dim3 g1024(DFF / 128, M / 128);
        gemm_bf16_kernel<<<g1024, 256, 0, stream>>>(Sbf, W1_b, b1, nullptr, nullptr, Hbf, M, DFF, Cd, 1);
        gemm_bf16_kernel<<<g256, 256, 0, stream>>>(Hbf, W2_b, b2, S32, T32, nullptr, M, Cd, DFF, 0);

        // LN2 -> X (f32 + bf16) and XQK = bf16(X + pe) for next pass
        ln_kernel<<<M, 256, 0, stream>>>(T32, g2, be2, X32, Xbf, XQK, pe, Nvhw);
    }

    // scatter back + residual, and mask
    final_kernel<<<M, 256, 0, stream>>>(backbone, X32, (float*)d_out);
    mask_kernel<<<1, 128, 0, stream>>>(defe, (float*)d_out + (size_t)2 * 256 * 128 * 128);
}
